// DenseVLADModel_36558761624566
// MI455X (gfx1250) — matrix operand order repaired and checked by reading, not compiled, not run on hardware
//
#include <hip/hip_runtime.h>
#include <hip/hip_bf16.h>
#include <math.h>

// ---------------------------------------------------------------------------
// DenseVLAD on MI455X (gfx1250, wave32).
//   feat : [64, 768, 17, 17] f32  -> vw [B=64, N=3468, D=64]
//   codes: [248, 64] f32
//   out  : [64, 248*64] f32
// Bandwidth-bound (~120 MB total traffic). WMMA (bf16 16x16x32) for the
// assignment-scoring GEMM (feeds a discrete argmin only); residual math f32.
// ---------------------------------------------------------------------------

typedef __attribute__((ext_vector_type(16))) __bf16 v16bf;
typedef __attribute__((ext_vector_type(8)))  float  v8f;

#define B_SZ   64
#define N_VW   3468          // 768*17*17 / 64
#define D_SZ   64
#define K_SZ   248
#define KPAD   256
#define TSTR   68            // LDS row stride (floats): float4-aligned, bank-spread
#define NSPLIT 8             // parallel splits of the N-axis column-norm reduction

// Order-preserving float -> uint transform (monotonic for all finite values).
__device__ __forceinline__ unsigned order_f32(float f) {
  unsigned u = __float_as_uint(f);
  return u ^ (unsigned)(((int)u >> 31) | 0x80000000);
}

// ---------------------------------------------------------------------------
// Kernel 1a: partial column sums of squares over an N-slice, atomic-accumulated
// into sums[b*64+d] (zeroed beforehand). grid=(B, NSPLIT), block=256.
// ---------------------------------------------------------------------------
__global__ __launch_bounds__(256) void vlad_colnorm_part(const float* __restrict__ feat,
                                                         float* __restrict__ sums) {
  const int b = blockIdx.x;
  const int split = blockIdx.y;
  const int tid = threadIdx.x;
  const int d = tid & 63;        // descriptor dim
  const int c = tid >> 6;        // 4 n-subchunks
  const int chunk = (N_VW + NSPLIT - 1) / NSPLIT;       // 434
  const int lo = split * chunk;
  const int hi = min(lo + chunk, N_VW);
  const float* fb = feat + (size_t)b * (N_VW * D_SZ);
  float s = 0.f;
  for (int n = lo + c; n < hi; n += 4) {
    float v = fb[(size_t)n * D_SZ + d];   // 64 consecutive lanes -> 256B coalesced
    s += v * v;
  }
  __shared__ float sh[256];
  sh[tid] = s;
  __syncthreads();
  if (tid < 64) {
    float t = sh[tid] + sh[tid + 64] + sh[tid + 128] + sh[tid + 192];
    atomicAdd(&sums[b * 64 + tid], t);
  }
}

// Kernel 1b: sums -> inverse norms, in place. grid=16, block=256.
__global__ void vlad_colnorm_fin(float* __restrict__ invnorm) {
  int i = blockIdx.x * blockDim.x + threadIdx.x;
  if (i < B_SZ * 64) invnorm[i] = 1.0f / fmaxf(sqrtf(invnorm[i]), 1e-12f);
}

// ---------------------------------------------------------------------------
// Kernel 2: codebook prep — squared norms (+huge pad) and bf16 conversion
// ---------------------------------------------------------------------------
__global__ __launch_bounds__(256) void vlad_prep(const float* __restrict__ codes,
                                                 float* __restrict__ cnorm2,
                                                 __bf16* __restrict__ codes_bf) {
  const int k = threadIdx.x;     // 0..255
  if (k < K_SZ) {
    float s = 0.f;
    for (int d = 0; d < D_SZ; ++d) { float v = codes[k * D_SZ + d]; s += v * v; }
    cnorm2[k] = s;
  } else {
    cnorm2[k] = 3.0e38f;         // padded codes never win argmin
  }
  for (int i = threadIdx.x; i < KPAD * D_SZ; i += 256) {
    int kk = i >> 6;
    codes_bf[i] = (kk < K_SZ) ? (__bf16)codes[i] : (__bf16)0.0f;
  }
}

// ---------------------------------------------------------------------------
// Kernel 3: zero a float buffer (per-launch deterministic accumulators)
// ---------------------------------------------------------------------------
__global__ void vlad_zero(float* __restrict__ p, int n) {
  int i = blockIdx.x * blockDim.x + threadIdx.x;
  if (i < n) p[i] = 0.f;
}

// ---------------------------------------------------------------------------
// Kernel 4: main — WMMA scoring, packed-u64 argmin, residual scatter-add.
// Block = 128 (4 waves); each wave owns one 16-row tile of n.
// ---------------------------------------------------------------------------
__global__ __launch_bounds__(128) void vlad_main(const float* __restrict__ feat,
                                                 const float* __restrict__ codes,
                                                 const float* __restrict__ invnorm,
                                                 const float* __restrict__ cnorm2,
                                                 const __bf16* __restrict__ codes_bf,
                                                 float* __restrict__ vlad) {
  __shared__ alignas(16) float s_tile[4][16 * TSTR];   // 4 waves x 16x64 f32 (+pad)
  __shared__ float s_inv[64];
  __shared__ float s_cn[KPAD];

  const int b    = blockIdx.y;
  const int tid  = threadIdx.x;
  const int wave = tid >> 5;
  const int lane = tid & 31;

  if (tid < 64) s_inv[tid] = invnorm[b * 64 + tid];
  for (int i = tid; i < KPAD; i += 128) s_cn[i] = cnorm2[i];

  const int n0 = (blockIdx.x * 4 + wave) * 16;
  const float* featb = feat + (size_t)b * (N_VW * D_SZ);
  float* tile = s_tile[wave];

  // ---- stage 16x64 f32 tile into LDS (coalesced float4), zero OOB rows ----
  #pragma unroll
  for (int j = 0; j < 8; ++j) {
    int idx = j * 32 + lane;     // 256 float4 slots
    int row = idx >> 4;
    int c4  = idx & 15;
    int n   = n0 + row;
    float4 v = make_float4(0.f, 0.f, 0.f, 0.f);
    if (n < N_VW) v = *(const float4*)(featb + (size_t)n * D_SZ + c4 * 4);
    *(float4*)(tile + row * TSTR + c4 * 4) = v;
  }
  __syncthreads();

  // ---- build A fragments (16x32 bf16, contiguous-k layout), scaled by inv col-norms ----
  // lane L: row M = L&15; elem e: d = e + 16*(L>=16)
  const int m  = lane & 15;
  const int hi = lane >> 4;
  v16bf a0, a1;
  #pragma unroll
  for (int e = 0; e < 16; ++e) {
    int d = e + hi * 16;
    a0[e] = (__bf16)(tile[m * TSTR + d]      * s_inv[d]);
    a1[e] = (__bf16)(tile[m * TSTR + 32 + d] * s_inv[32 + d]);
  }

  // ---- score all 256 (padded) codes: 16 tiles x 2 WMMAs; running argmin ----
  // Packed key: (orderable(dist_key) << 32) | k  => u64-min == argmin, ties->low k
  unsigned long long best[8];
  #pragma unroll
  for (int i = 0; i < 8; ++i) best[i] = ~0ull;

  const int kcol = lane & 15;
  for (int kt = 0; kt < 16; ++kt) {
    // B (32x16 bf16): lane L holds column n=L&15, 16 contiguous d from 16*(L>=16)
    const __bf16* brow = codes_bf + ((kt * 16 + kcol) * D_SZ + hi * 16);
    v16bf b0 = *(const v16bf*)(brow);        // d in [hi*16, hi*16+16)
    v16bf b1 = *(const v16bf*)(brow + 32);   // d in [32+hi*16, ...)
    v8f c = {0.f, 0.f, 0.f, 0.f, 0.f, 0.f, 0.f, 0.f};
    c = __builtin_amdgcn_wmma_f32_16x16x32_bf16(false, a0, false, b0,
                                                (short)0, c, false, false);
    c = __builtin_amdgcn_wmma_f32_16x16x32_bf16(false, a1, false, b1,
                                                (short)0, c, false, false);
    const unsigned kk = (unsigned)(kt * 16 + kcol);
    const float cn = s_cn[kk];
    #pragma unroll
    for (int i = 0; i < 8; ++i) {            // C: VGPR i -> row i + 8*hi
      float key = cn - 2.0f * c[i];          // ||vw||^2 term is constant per row
      unsigned long long p = ((unsigned long long)order_f32(key) << 32) | kk;
      best[i] = (p < best[i]) ? p : best[i];
    }
  }

  // ---- argmin across the 16 code-columns of each half (u64 xor-min) ----
  #pragma unroll
  for (int mask = 1; mask <= 8; mask <<= 1) {
    #pragma unroll
    for (int i = 0; i < 8; ++i) {
      unsigned long long o = __shfl_xor(best[i], mask, 32);
      best[i] = (o < best[i]) ? o : best[i];
    }
  }

  // ---- residuals: r = vw_norm - code[k*], row-normalized, atomic scatter ----
  const int d0 = lane, d1 = lane + 32;
  auto scatter = [&](int row, int kstar) {
    if (n0 + row >= N_VW) return;
    float v0 = tile[row * TSTR + d0] * s_inv[d0];
    float v1 = tile[row * TSTR + d1] * s_inv[d1];
    const float* crow = codes + (size_t)kstar * D_SZ;
    float r0 = v0 - crow[d0];
    float r1 = v1 - crow[d1];
    float ss = r0 * r0 + r1 * r1;
    #pragma unroll
    for (int mask = 1; mask < 32; mask <<= 1) ss += __shfl_xor(ss, mask, 32);
    float inv = 1.0f / (sqrtf(ss) + 1e-8f);
    float* vrow = vlad + ((size_t)b * K_SZ + kstar) * D_SZ;
    atomicAdd(vrow + d0, r0 * inv);
    atomicAdd(vrow + d1, r1 * inv);
  };

  #pragma unroll
  for (int i = 0; i < 8; ++i) {
    int kA = (int)(unsigned)__shfl(best[i], 0, 32);    // row i     (lanes 0..15)
    int kB = (int)(unsigned)__shfl(best[i], 16, 32);   // row i + 8 (lanes 16..31)
    scatter(i,     kA);
    scatter(i + 8, kB);
  }
}

// ---------------------------------------------------------------------------
// Kernel 5: per-batch-row standardization (unbiased std, torch semantics)
// ---------------------------------------------------------------------------
__global__ __launch_bounds__(256) void vlad_norm(const float* __restrict__ vlad,
                                                 float* __restrict__ out) {
  const int b = blockIdx.x;
  const int tid = threadIdx.x;
  const int M = K_SZ * D_SZ;   // 15872
  const float* row = vlad + (size_t)b * M;
  float s = 0.f, q = 0.f;
  for (int i = tid; i < M; i += 256) { float v = row[i]; s += v; q += v * v; }
  __shared__ float sh_s[256], sh_q[256];
  sh_s[tid] = s; sh_q[tid] = q;
  __syncthreads();
  for (int o = 128; o > 0; o >>= 1) {
    if (tid < o) { sh_s[tid] += sh_s[tid + o]; sh_q[tid] += sh_q[tid + o]; }
    __syncthreads();
  }
  float sum = sh_s[0], sq = sh_q[0];
  float mean = sum / (float)M;
  float var  = (sq - sum * mean) / (float)(M - 1);
  float denom = sqrtf(fmaxf(var, 0.f)) + 1e-8f;
  float* orow = out + (size_t)b * M;
  for (int i = tid; i < M; i += 256) orow[i] = (row[i] - mean) / denom;
}

// ---------------------------------------------------------------------------
extern "C" void kernel_launch(void* const* d_in, const int* in_sizes, int n_in,
                              void* d_out, int out_size, void* d_ws, size_t ws_size,
                              hipStream_t stream) {
  (void)in_sizes; (void)n_in; (void)out_size; (void)ws_size;
  const float* feat  = (const float*)d_in[0];
  const float* codes = (const float*)d_in[1];

  // workspace layout (all offsets 32B-aligned)
  char* ws = (char*)d_ws;
  float*  invnorm  = (float*)(ws);                          // 64*64*4   = 16384
  float*  cnorm2   = (float*)(ws + 16384);                  // 256*4     = 1024
  __bf16* codes_bf = (__bf16*)(ws + 16384 + 1024);          // 256*64*2  = 32768
  float*  vlad     = (float*)(ws + 16384 + 1024 + 32768);   // 64*248*64*4 ~ 4 MB
  float*  out      = (float*)d_out;

  // zero the atomic accumulators (invnorm doubles as the sums buffer)
  vlad_zero<<<16, 256, 0, stream>>>(invnorm, B_SZ * 64);
  const int nz = B_SZ * K_SZ * D_SZ;
  vlad_zero<<<(nz + 255) / 256, 256, 0, stream>>>(vlad, nz);

  dim3 cn_grid(B_SZ, NSPLIT);   // 512 blocks: wide enough to stream 23.3 TB/s
  vlad_colnorm_part<<<cn_grid, 256, 0, stream>>>(feat, invnorm);
  vlad_colnorm_fin<<<16, 256, 0, stream>>>(invnorm);
  vlad_prep<<<1, 256, 0, stream>>>(codes, cnorm2, codes_bf);

  dim3 grid(55, B_SZ);          // 55 blocks * 4 waves * 16 rows = 3520 >= N_VW
  vlad_main<<<grid, 128, 0, stream>>>(feat, codes, invnorm, cnorm2, codes_bf, vlad);
  vlad_norm<<<B_SZ, 256, 0, stream>>>(vlad, out);
}
